// Fasttext_300_1486058684815
// MI455X (gfx1250) — matrix operand order repaired and checked by reading, not compiled, not run on hardware
//
#include <hip/hip_runtime.h>
#include <hip/hip_bf16.h>
#include <math.h>

typedef __attribute__((ext_vector_type(16))) _Float16 v16h;
typedef __attribute__((ext_vector_type(8)))  float    v8f;

#define D_FEAT 300

// ---------------------------------------------------------------------------
// CSR build step 1: in-degree histogram (int atomics, 800K ops -- cheap)
// ---------------------------------------------------------------------------
__global__ __launch_bounds__(256) void degree_kernel(
    const int* __restrict__ dst, int* __restrict__ deg, int n_edges)
{
  int e = blockIdx.x * 256 + threadIdx.x;
  if (e < n_edges) atomicAdd(&deg[dst[e]], 1);
}

// ---------------------------------------------------------------------------
// CSR build step 2: exclusive prefix scan over degrees (single block, 2-level:
// per-thread serial slice sums -> LDS scan -> per-thread prefix writeback).
// Writes rowptr[0..n] and initializes cursor[] = rowptr[].
// ---------------------------------------------------------------------------
__global__ __launch_bounds__(256) void exclusive_scan_kernel(
    const int* __restrict__ deg, int* __restrict__ rowptr,
    int* __restrict__ cursor, int n)
{
  __shared__ int totals[256];
  const int tid = threadIdx.x;
  const int per = (n + 255) / 256;
  const int lo = tid * per;
  const int hi = (lo + per < n) ? (lo + per) : n;

  int s = 0;
  for (int i = lo; i < hi; ++i) s += deg[i];
  totals[tid] = s;
  __syncthreads();

  if (tid == 0) {
    int run = 0;
    for (int i = 0; i < 256; ++i) { int t = totals[i]; totals[i] = run; run += t; }
    rowptr[n] = run;                       // total edge count
  }
  __syncthreads();

  int run = totals[tid];
  for (int i = lo; i < hi; ++i) {
    rowptr[i] = run;
    cursor[i] = run;
    run += deg[i];
  }
}

// ---------------------------------------------------------------------------
// CSR build step 3: adj[pos] = src[e], pos = cursor[dst[e]]++  (int atomics)
// ---------------------------------------------------------------------------
__global__ __launch_bounds__(256) void fill_adj_kernel(
    const int* __restrict__ src, const int* __restrict__ dst,
    int* __restrict__ cursor, int* __restrict__ adj, int n_edges)
{
  int e = blockIdx.x * 256 + threadIdx.x;
  if (e < n_edges) {
    int p = atomicAdd(&cursor[dst[e]], 1);
    adj[p] = src[e];
  }
}

// ---------------------------------------------------------------------------
// Pull-mode aggregation, fused with f16 pack:
//   h[node, 300 + col] = (f16) sum_{u in in(node)} x[u, col]
// Plain loads + register accumulation; zero float atomics.  Rows are 1200B
// contiguous and L2-resident (60MB dataset vs 192MB L2).
// ---------------------------------------------------------------------------
__global__ __launch_bounds__(256) void gather_h_kernel(
    const float* __restrict__ x, const int* __restrict__ rowptr,
    const int* __restrict__ adj, _Float16* __restrict__ h, int kpad)
{
  int node = blockIdx.y;
  int col = blockIdx.x * blockDim.x + threadIdx.x;
  if (col >= D_FEAT) return;
  int beg = rowptr[node], end = rowptr[node + 1];
  float s = 0.f;
  for (int i = beg; i < end; ++i)
    s += x[(size_t)adj[i] * D_FEAT + col];
  h[(size_t)node * kpad + D_FEAT + col] = (_Float16)s;
}

// ---------------------------------------------------------------------------
// Fill self-feature half + zero K-padding of h:
//   cols [0,300) = x ; if skip_mid: leave [300,600) for gather_h ; rest = 0
// ---------------------------------------------------------------------------
__global__ __launch_bounds__(256) void build_h_kernel(
    const float* __restrict__ x, _Float16* __restrict__ h, int kpad, int skip_mid)
{
  int node = blockIdx.y;
  int c = blockIdx.x * blockDim.x + threadIdx.x;
  if (c >= kpad) return;
  if (skip_mid && c >= D_FEAT && c < 2 * D_FEAT) return;   // gather_h owns these
  float v = (c < D_FEAT) ? x[(size_t)node * D_FEAT + c] : 0.f;
  h[(size_t)node * kpad + c] = (_Float16)v;
}

// ---------------------------------------------------------------------------
// Wt[k, n] = W[n, k] as f16, zero padded to [kpad, npad].  W is [300, K].
// ---------------------------------------------------------------------------
__global__ __launch_bounds__(256) void transpose_w_kernel(
    const float* __restrict__ W, _Float16* __restrict__ Wt,
    int K, int kpad, int npad)
{
  int i = blockIdx.x * blockDim.x + threadIdx.x;
  if (i >= kpad * npad) return;
  int k = i / npad;
  int n = i % npad;
  float v = (k < K && n < D_FEAT) ? W[(size_t)n * K + k] : 0.f;
  Wt[i] = (_Float16)v;
}

// ---------------------------------------------------------------------------
// WMMA GEMM:  Out[M, 300] = act( H[M, KPAD] @ Wt[KPAD, 320] + bias )
// One wave per block; 16(M) x 64(N) strip, 4 accumulators sharing one A frag.
// N padded to 320 (20 tiles, grid.y=5): the K-loop is branch-free, EXEC stays
// all-ones, accumulators stay pinned, loads clause-batch against v_wmma.
// ACT: 0 identity, 1 leaky_relu(0.01), 2 tanh
// ---------------------------------------------------------------------------
template <int KPAD, int ACT>
__global__ __launch_bounds__(32) void gemm_wmma_kernel(
    const _Float16* __restrict__ H, const _Float16* __restrict__ Wt,
    const float* __restrict__ bias, float* __restrict__ Out)
{
  constexpr int NPW = 320;
  const int lane  = threadIdx.x;
  const int mtile = blockIdx.x;
  const int nt0   = blockIdx.y * 4;

  const int arow  = mtile * 16 + (lane & 15);
  const int akoff = (lane & 16) ? 8 : 0;
  const _Float16* __restrict__ hrow  = H + (size_t)arow * KPAD + akoff;
  const _Float16* __restrict__ wbase = Wt + nt0 * 16 + (lane & 15);

  v8f acc0 = {}, acc1 = {}, acc2 = {}, acc3 = {};

  #pragma unroll
  for (int ks = 0; ks < KPAD; ks += 32) {
    v16h a;
    {
      uint4* ap = (uint4*)&a;
      ap[0] = *(const uint4*)(hrow + ks);        // K = ks+akoff    .. +7
      ap[1] = *(const uint4*)(hrow + ks + 16);   // K = ks+akoff+16 .. +23
    }
    const _Float16* __restrict__ wrow = wbase + (size_t)ks * NPW;
    v16h b0, b1, b2, b3;
    {
      #pragma unroll
      for (int s = 0; s < 16; ++s) {
        const size_t kk = (size_t)(akoff + (s < 8 ? s : s + 8)) * NPW;
        b0[s] = wrow[kk +  0]; b1[s] = wrow[kk + 16];
        b2[s] = wrow[kk + 32]; b3[s] = wrow[kk + 48];
      }
    }
    acc0 = __builtin_amdgcn_wmma_f32_16x16x32_f16(false, a, false, b0, (short)0, acc0, false, false);
    acc1 = __builtin_amdgcn_wmma_f32_16x16x32_f16(false, a, false, b1, (short)0, acc1, false, false);
    acc2 = __builtin_amdgcn_wmma_f32_16x16x32_f16(false, a, false, b2, (short)0, acc2, false, false);
    acc3 = __builtin_amdgcn_wmma_f32_16x16x32_f16(false, a, false, b3, (short)0, acc3, false, false);
  }

  // ---- epilogue: bias + activation + masked store (divergence OK here) ----
  const int rbase = mtile * 16 + ((lane & 16) ? 8 : 0);
  const int cbase = nt0 * 16 + (lane & 15);
  v8f* accs[4] = { &acc0, &acc1, &acc2, &acc3 };
  #pragma unroll
  for (int nt = 0; nt < 4; ++nt) {
    int col = cbase + nt * 16;
    if (col < D_FEAT) {
      float bv = bias[col];
      #pragma unroll
      for (int r = 0; r < 8; ++r) {
        float v = (*accs[nt])[r] + bv;
        if (ACT == 1)      v = (v > 0.f) ? v : 0.01f * v;
        else if (ACT == 2) v = tanhf(v);
        Out[(size_t)(rbase + r) * D_FEAT + col] = v;
      }
    }
  }
}

// ---------------------------------------------------------------------------
// Global sum of squares (Frobenius norm), wave-reduced + atomic.
// ---------------------------------------------------------------------------
__global__ __launch_bounds__(256) void sumsq_kernel(
    const float* __restrict__ y, float* __restrict__ s, size_t n)
{
  size_t stride = (size_t)gridDim.x * blockDim.x;
  float v = 0.f;
  for (size_t i = (size_t)blockIdx.x * blockDim.x + threadIdx.x; i < n; i += stride)
    v += y[i] * y[i];
  for (int off = 16; off > 0; off >>= 1)
    v += __shfl_down(v, off, 32);
  if ((threadIdx.x & 31) == 0)
    atomicAdd(s, v);
}

__global__ __launch_bounds__(256) void scale_kernel(
    const float* __restrict__ y, const float* __restrict__ s,
    float* __restrict__ out, size_t n)
{
  size_t i = (size_t)blockIdx.x * blockDim.x + threadIdx.x;
  if (i >= n) return;
  out[i] = y[i] * rsqrtf(*s);
}

// ---------------------------------------------------------------------------
// Launcher
// ---------------------------------------------------------------------------
extern "C" void kernel_launch(void* const* d_in, const int* in_sizes, int n_in,
                              void* d_out, int out_size, void* d_ws, size_t ws_size,
                              hipStream_t stream) {
  const float* feat = (const float*)d_in[0];
  const int*   src  = (const int*)  d_in[1];
  const int*   dst  = (const int*)  d_in[2];
  const float* W1   = (const float*)d_in[3];
  const float* b1   = (const float*)d_in[4];
  const float* W2   = (const float*)d_in[5];
  const float* b2   = (const float*)d_in[6];
  const float* W3   = (const float*)d_in[7];
  const float* b3   = (const float*)d_in[8];
  float* out = (float*)d_out;

  const int n_nodes = in_sizes[0] / D_FEAT;   // 50000 (divisible by 16)
  const int n_edges = in_sizes[1];            // 800000

  constexpr int KP12 = 608;   // 600 padded to multiple of 32
  constexpr int KP3  = 320;   // 300 padded to multiple of 32
  constexpr int NP   = 320;   // 300 padded to 20 WMMA tiles (branch-free loop)

  // ---- workspace layout (16B aligned offsets) ----
  char* ws = (char*)d_ws;
  float*    xbuf   = (float*)   (ws + 0);                     //  60,000,000 B
  float*    x2buf  = (float*)   (ws + 60000000);              //  60,000,000 B
  _Float16* h      = (_Float16*)(ws + 120000000);             //  60,800,000 B
  _Float16* wt1    = (_Float16*)(ws + 180800000);             //     389,120 B
  _Float16* wt2    = (_Float16*)(ws + 181189120);             //     389,120 B
  _Float16* wt3    = (_Float16*)(ws + 181578240);             //     204,800 B
  float*    ssq    = (float*)   (ws + 181783040);             //          16 B
  int*      deg    = (int*)     (ws + 181783056);             //     200,000 B
  int*      rowptr = (int*)     (ws + 181983056);             //     200,016 B
  int*      cursor = (int*)     (ws + 182183072);             //     200,000 B
  int*      adj    = (int*)     (ws + 182383072);             //   3,200,000 B

  const size_t nOut = (size_t)n_nodes * D_FEAT;

  dim3 blk256(256);
  dim3 gemmBlk(32);
  dim3 gemmGrid(n_nodes / 16, (NP / 16) / 4);      // 3125 x 5
  dim3 hGrid12((KP12 + 255) / 256, n_nodes);
  dim3 hGrid3 ((KP3  + 255) / 256, n_nodes);
  dim3 gGrid  ((D_FEAT + 255) / 256, n_nodes);     // 2 x 50000
  int  eBlocks = (n_edges + 255) / 256;

  // ---- CSR by destination (built once, reused by both GCN layers) ----
  hipMemsetAsync(deg, 0, (size_t)n_nodes * sizeof(int), stream);
  degree_kernel<<<eBlocks, blk256, 0, stream>>>(dst, deg, n_edges);
  exclusive_scan_kernel<<<1, blk256, 0, stream>>>(deg, rowptr, cursor, n_nodes);
  fill_adj_kernel<<<eBlocks, blk256, 0, stream>>>(src, dst, cursor, adj, n_edges);

  // ---- weight transposes (f32 -> padded f16, B-layout) ----
  transpose_w_kernel<<<(KP12 * NP + 255) / 256, blk256, 0, stream>>>(W1, wt1, 600, KP12, NP);
  transpose_w_kernel<<<(KP12 * NP + 255) / 256, blk256, 0, stream>>>(W2, wt2, 600, KP12, NP);
  transpose_w_kernel<<<(KP3  * NP + 255) / 256, blk256, 0, stream>>>(W3, wt3, 300, KP3,  NP);

  // ---- layer 1:  x1 = leaky_relu([feat | agg(feat)] @ W1^T + b1) ----
  build_h_kernel <<<hGrid12, blk256, 0, stream>>>(feat, h, KP12, 1);
  gather_h_kernel<<<gGrid,   blk256, 0, stream>>>(feat, rowptr, adj, h, KP12);
  gemm_wmma_kernel<KP12, 1><<<gemmGrid, gemmBlk, 0, stream>>>(h, wt1, b1, xbuf);

  // ---- layer 2:  x2 = [x1 | agg(x1)] @ W2^T + b2 ----
  build_h_kernel <<<hGrid12, blk256, 0, stream>>>(xbuf, h, KP12, 1);
  gather_h_kernel<<<gGrid,   blk256, 0, stream>>>(xbuf, rowptr, adj, h, KP12);
  gemm_wmma_kernel<KP12, 0><<<gemmGrid, gemmBlk, 0, stream>>>(h, wt2, b2, x2buf);

  // ---- layer 3:  y = tanh(x2 @ W3^T + b3)   (y -> xbuf) ----
  build_h_kernel<<<hGrid3, blk256, 0, stream>>>(x2buf, h, KP3, 0);
  gemm_wmma_kernel<KP3, 2><<<gemmGrid, gemmBlk, 0, stream>>>(h, wt3, b3, xbuf);

  // ---- global Frobenius-norm scaling ----
  hipMemsetAsync(ssq, 0, sizeof(float), stream);
  sumsq_kernel<<<4096, blk256, 0, stream>>>(xbuf, ssq, nOut);
  scale_kernel<<<(unsigned)((nOut + 255) / 256), blk256, 0, stream>>>(xbuf, ssq, out, nOut);
}
